// HomogeneousGATMultiTask_12592844112466
// MI455X (gfx1250) — matrix operand order repaired and checked by reading, not compiled, not run on hardware
//
#include <hip/hip_runtime.h>

#define HEADS  4
#define NNODES 100000
#define NEDGES 800000
#define NLAB   200000

typedef __attribute__((ext_vector_type(16))) _Float16     v16h;
typedef __attribute__((ext_vector_type(8)))  _Float16     v8h;
typedef __attribute__((ext_vector_type(4)))  _Float16     v4h;
typedef __attribute__((ext_vector_type(8)))  float        v8f;
typedef __attribute__((ext_vector_type(4)))  float        v4f;
typedef __attribute__((ext_vector_type(4)))  unsigned int v4u;
typedef __attribute__((ext_vector_type(8)))  int          v8i;
typedef __attribute__((ext_vector_type(4)))  int          v4i;

// ---------------- float atomic max via int/uint trick ----------------
__device__ __forceinline__ void atomicMaxF(float* addr, float val) {
    if (val >= 0.0f) atomicMax((int*)addr, __float_as_int(val));
    else             atomicMin((unsigned int*)addr, __float_as_uint(val));
}

// ---------------- generic fill ----------------
__global__ void fill_kernel(float* __restrict__ p, long n, float v) {
    long i = (long)blockIdx.x * blockDim.x + threadIdx.x;
    if (i < n) p[i] = v;
}

// =====================================================================
// Weight pre-conversion: B[K,M] f32 -> padded fragment-ordered f16.
// Chunk c (one per GEMM blockIdx.y) is a contiguous K32*8*512-f16 span:
//   frag (kt,lmt) at (kt*8+lmt)*512, lane-major, 16 contiguous f16/lane,
//   element o of lane l = B[kt*32 + ((l>>4)<<4) + o][(c*8+lmt)*16 + (l&15)]
// Out-of-range columns padded with zeros.
// =====================================================================
__global__ void wfrag_kernel(const float* __restrict__ B, _Float16* __restrict__ out,
                             int K32, int M, int total) {
    int idx = blockIdx.x * blockDim.x + threadIdx.x;
    if (idx >= total) return;
    int cs    = K32 << 12;                 // chunk size in f16
    int chunk = idx / cs;
    int rem   = idx - chunk * cs;
    int o  = rem & 15;
    int l  = (rem >> 4) & 31;
    int f  = rem >> 9;                     // kt*8 + lmt (padded stride 8)
    int kt = f >> 3, lmt = f & 7;
    int k  = kt * 32 + ((l >> 4) << 4) + o;
    int m  = (chunk * 8 + lmt) * 16 + (l & 15);
    out[idx] = (m < M) ? (_Float16)B[(size_t)k * M + m] : (_Float16)0.0f;
}

// ---- Tensor Data Mover: 1-D DMA of nElem f16 from global to LDS ----
// D# per CDNA5 ISA ch.8: group0 = {count=1, lds_addr, global_addr, type=2},
// group1 = {data_size=2B, tensor_dim0=nElem, tensor_dim1=1, tile_dim0=nElem,
//           tile_dim1=0 (1-D), stride0=nElem}. Groups 2/3 unused (<=2D).
// clang-23 toolchain form: 6 args (extra int32x8 group before cpol).
__device__ __forceinline__ void tdm_load_1d(const _Float16* gsrc, unsigned ldsOff,
                                            int nElem) {
    unsigned long long ga = (unsigned long long)(size_t)gsrc;
    v4u g0 = { 1u,                                         // count=1, user mode
               ldsOff,                                     // lds_addr [63:32]
               (unsigned)(ga & 0xFFFFFFFFu),               // global_addr lo
               (unsigned)((ga >> 32) & 0x1FFFFFFu) | (2u << 30) };  // ga hi | type=2
    unsigned ne = (unsigned)nElem;
    v8i g1 = { (int)(1u << 16),                            // data_size=1 (2 bytes)
               (int)((ne & 0xFFFFu) << 16),                // tensor_dim0[15:0]
               (int)((ne >> 16) | (1u << 16)),             // tensor_dim0[31:16], tensor_dim1=1
               (int)((ne & 0xFFFFu) << 16),                // tile_dim0[15:0]
               0,                                          // tile_dim1=0, tile_dim2=0
               (int)ne,                                    // tensor_dim0_stride[31:0]
               0, 0 };
    v4i g2 = { 0, 0, 0, 0 };
    v4i g3 = { 0, 0, 0, 0 };
    v8i g4 = { 0, 0, 0, 0, 0, 0, 0, 0 };
    __builtin_amdgcn_tensor_load_to_lds(g0, g1, g2, g3, g4, 0);
}

// =====================================================================
// WMMA GEMM: C[rows,M] = A[rows,K] * B[K,M] (+bias, optional relu).
// Block = 256 threads = 8 waves; 128-row strip x 8 M-tile chunk.
//  - B chunk arrives via ONE tensor_load_to_lds DMA (pre-converted f16
//    fragment buffer), overlapped with A staging; s_wait_tensorcnt + barrier.
//  - A strip staged f32->f16 into LDS with vectorized loads; each wave
//    keeps its K32 A fragments in registers across all M-tiles.
// Hot loop: 2x ds_load_b128 + v_wmma_f32_16x16x32_f16 per (M-tile, k-step).
// LDS = K32 * 16 KB.
// =====================================================================
#define MT_PER_BLOCK 8

template <int K32>
__global__ void gemm_wmma_kernel(const _Float16* __restrict__ Bfrag,
                                 const float* __restrict__ A,
                                 const float* __restrict__ bias,
                                 float* __restrict__ C,
                                 int rows, int M, int relu) {
    constexpr int K = K32 * 32;
    constexpr int CHUNK = K32 * MT_PER_BLOCK * 512;       // f16 per B chunk
    extern __shared__ __align__(16) _Float16 smem[];
    _Float16* sB = smem;                                  // CHUNK f16
    _Float16* sA = smem + CHUNK;                          // 128*K f16

    const int tilesM = (M + 15) >> 4;
    const int mt0    = blockIdx.y * MT_PER_BLOCK;
    const int nMt    = min(MT_PER_BLOCK, tilesM - mt0);
    const int row0   = blockIdx.x * 128;
    const int tid    = threadIdx.x;

    // ---- kick off B-chunk DMA from wave 0 (TENSORcnt, EXEC-independent) ----
    if (tid < 32) {
        tdm_load_1d(Bfrag + (size_t)blockIdx.y * CHUNK,
                    (unsigned)(size_t)sB, CHUNK);
    }

    // ---- stage A strip into LDS (row-major f16 [128][K]), 4-wide ----
    for (int idx4 = tid; idx4 < 32 * K; idx4 += 256) {    // 128*K/4 groups
        int base = idx4 << 2;
        int rr = base / K, kk = base - rr * K;
        int grow = row0 + rr;
        v4f xv = {};
        if (grow < rows) xv = *(const v4f*)(A + (size_t)grow * K + kk);
        v4h hv = { (_Float16)xv.x, (_Float16)xv.y, (_Float16)xv.z, (_Float16)xv.w };
        *(v4h*)(sA + base) = hv;
    }

    if (tid < 32) __builtin_amdgcn_s_wait_tensorcnt(0);
    __syncthreads();

    // ---- per-wave compute: one 16-row tile, all nMt column tiles ----
    const int wave = tid >> 5, lane = tid & 31;
    const int r   = lane & 15;            // A row within tile
    const int ka  = (lane >> 4) << 4;     // A k-base (0/16) within 32-step
    const int n   = lane & 15;            // B/C col within tile
    const int mhi = (lane >> 4) << 3;     // C row-half offset
    const int growBase = row0 + wave * 16;
    if (growBase >= rows) return;
    const bool fullTile = (growBase + 16) <= rows;

    // A fragments -> registers (two ds_load_b128 each)
    v16h af[K32];
    #pragma unroll
    for (int ks = 0; ks < K32; ++ks) {
        const _Float16* p = sA + (wave * 16 + r) * K + ks * 32 + ka;
        v8h lo = *(const v8h*)p;
        v8h hi = *(const v8h*)(p + 8);
        af[ks] = __builtin_shufflevector(lo, hi, 0, 1, 2, 3, 4, 5, 6, 7,
                                                 8, 9, 10, 11, 12, 13, 14, 15);
    }

    for (int lmt = 0; lmt < nMt; ++lmt) {
        v8f acc = {};
        #pragma unroll
        for (int ks = 0; ks < K32; ++ks) {
            const _Float16* q = sB + ((ks * MT_PER_BLOCK + lmt) * 32 + lane) * 16;
            v8h lo = *(const v8h*)q;
            v8h hi = *(const v8h*)(q + 8);
            v16h b = __builtin_shufflevector(lo, hi, 0, 1, 2, 3, 4, 5, 6, 7,
                                                     8, 9, 10, 11, 12, 13, 14, 15);
            acc = __builtin_amdgcn_wmma_f32_16x16x32_f16(false, af[ks], false, b,
                                                         (short)0, acc, false, false);
        }
        int bcol = (mt0 + lmt) * 16 + n;
        if (bcol < M) {
            float bv = bias ? bias[bcol] : 0.0f;
            float* cp = C + (size_t)growBase * M + bcol;
            if (fullTile) {
                #pragma unroll
                for (int v = 0; v < 8; ++v) {
                    float o = acc[v] + bv;
                    if (relu) o = fmaxf(o, 0.0f);
                    cp[(size_t)(mhi + v) * M] = o;
                }
            } else {
                #pragma unroll
                for (int v = 0; v < 8; ++v) {
                    if (growBase + mhi + v < rows) {
                        float o = acc[v] + bv;
                        if (relu) o = fmaxf(o, 0.0f);
                        cp[(size_t)(mhi + v) * M] = o;
                    }
                }
            }
        }
    }
}

// ---------------- attention coefficients: alpha_src / alpha_dst ----------------
__global__ void alpha_kernel(const float* __restrict__ h,
                             const float* __restrict__ as,
                             const float* __restrict__ ad,
                             float* __restrict__ alphaS,
                             float* __restrict__ alphaD, int C) {
    int idx = blockIdx.x * blockDim.x + threadIdx.x;
    if (idx >= NNODES * HEADS) return;
    int hd = idx & 3;
    int node = idx >> 2;
    const float* hp = h + (size_t)node * HEADS * C + hd * C;
    const float* s  = as + hd * C;
    const float* d  = ad + hd * C;
    float ss = 0.0f, dd = 0.0f;
    for (int c = 0; c < C; ++c) { float v = hp[c]; ss += v * s[c]; dd += v * d[c]; }
    alphaS[idx] = ss;
    alphaD[idx] = dd;
}

__device__ __forceinline__ float leaky02(float v) { return v > 0.0f ? v : 0.2f * v; }

// ---------------- edge pass 1: segment max ----------------
__global__ void edge_max_kernel(const int* __restrict__ src, const int* __restrict__ dst,
                                const float* __restrict__ aS, const float* __restrict__ aD,
                                float* __restrict__ m) {
    int idx = blockIdx.x * blockDim.x + threadIdx.x;
    if (idx >= NEDGES * HEADS) return;
    int e = idx >> 2, hd = idx & 3;
    int s = src[e], d = dst[e];
    float v = leaky02(aS[s * HEADS + hd] + aD[d * HEADS + hd]);
    atomicMaxF(&m[d * HEADS + hd], v);
}

// ---------------- edge pass 2: segment sum of exp ----------------
__global__ void edge_den_kernel(const int* __restrict__ src, const int* __restrict__ dst,
                                const float* __restrict__ aS, const float* __restrict__ aD,
                                const float* __restrict__ m, float* __restrict__ den) {
    int idx = blockIdx.x * blockDim.x + threadIdx.x;
    if (idx >= NEDGES * HEADS) return;
    int e = idx >> 2, hd = idx & 3;
    int s = src[e], d = dst[e];
    float v = leaky02(aS[s * HEADS + hd] + aD[d * HEADS + hd]);
    atomicAdd(&den[d * HEADS + hd], __expf(v - m[d * HEADS + hd]));
}

// ---------------- edge pass 3: weighted scatter-aggregate ----------------
__global__ void edge_agg_kernel(const int* __restrict__ src, const int* __restrict__ dst,
                                const float* __restrict__ aS, const float* __restrict__ aD,
                                const float* __restrict__ m, const float* __restrict__ den,
                                const float* __restrict__ h, float* __restrict__ agg, int C) {
    int idx = blockIdx.x * blockDim.x + threadIdx.x;
    if (idx >= NEDGES * HEADS) return;
    int e = idx >> 2, hd = idx & 3;
    int s = src[e], d = dst[e];
    float v = leaky02(aS[s * HEADS + hd] + aD[d * HEADS + hd]);
    float al = __expf(v - m[d * HEADS + hd]) / (den[d * HEADS + hd] + 1e-16f);
    const float* hp = h   + (size_t)s * HEADS * C + hd * C;
    float*       op = agg + (size_t)d * HEADS * C + hd * C;
    for (int c = 0; c < C; ++c) atomicAdd(&op[c], hp[c] * al);
}

// ---------------- mean over heads + bias + relu ----------------
__global__ void mean_kernel(const float* __restrict__ agg, const float* __restrict__ bias,
                            float* __restrict__ out, int C) {
    int idx = blockIdx.x * blockDim.x + threadIdx.x;
    if (idx >= NNODES * C) return;
    int node = idx / C, c = idx - node * C;
    const float* p = agg + (size_t)node * HEADS * C + c;
    float s = 0.0f;
    #pragma unroll
    for (int hd = 0; hd < HEADS; ++hd) s += p[hd * C];
    float o = s * 0.25f + bias[c];
    out[idx] = o > 0.0f ? o : 0.0f;
}

// ---------------- link decoder: per labeled edge dot(z[s], z[t]) ----------------
__global__ void link_kernel(const int* __restrict__ eli, const float* __restrict__ z,
                            float* __restrict__ out) {
    int i = blockIdx.x * blockDim.x + threadIdx.x;
    if (i >= NLAB) return;
    int s = eli[i], t = eli[NLAB + i];
    const float* zs = z + (size_t)s * 32;
    const float* zt = z + (size_t)t * 32;
    float acc = 0.0f;
    #pragma unroll
    for (int c = 0; c < 32; ++c) acc += zs[c] * zt[c];
    out[i] = acc;
}

extern "C" void kernel_launch(void* const* d_in, const int* in_sizes, int n_in,
                              void* d_out, int out_size, void* d_ws, size_t ws_size,
                              hipStream_t stream) {
    (void)in_sizes; (void)n_in; (void)out_size; (void)ws_size;
    const float* x    = (const float*)d_in[0];
    const int*   ei   = (const int*)d_in[1];
    const int*   eli  = (const int*)d_in[2];
    const float* W0   = (const float*)d_in[3];
    const float* as0  = (const float*)d_in[4];
    const float* ad0  = (const float*)d_in[5];
    const float* b0   = (const float*)d_in[6];
    const float* W1   = (const float*)d_in[7];
    const float* as1  = (const float*)d_in[8];
    const float* ad1  = (const float*)d_in[9];
    const float* b1   = (const float*)d_in[10];
    const float* linw = (const float*)d_in[11];
    const float* linb = (const float*)d_in[12];
    const float* d1w  = (const float*)d_in[13];
    const float* d1b  = (const float*)d_in[14];
    const float* d2w  = (const float*)d_in[15];
    const float* d2b  = (const float*)d_in[16];
    const float* d3w  = (const float*)d_in[17];
    const float* d3b  = (const float*)d_in[18];
    const float* d4w  = (const float*)d_in[19];
    const float* d4b  = (const float*)d_in[20];

    const int* src = ei;
    const int* dst = ei + NEDGES;

    float* ws = (float*)d_ws;
    size_t off = 0;
    float* bufA = ws + off; off += (size_t)NNODES * 256;  // h0 / h1 / z
    float* bufB = ws + off; off += (size_t)NNODES * 256;  // agg0 / agg1 / e2
    float* bufC = ws + off; off += (size_t)NNODES * 96;   // hr0 / hr1 / e1 / e3
    float* aS   = ws + off; off += (size_t)NNODES * HEADS;
    float* aD   = ws + off; off += (size_t)NNODES * HEADS;
    float* mMax = ws + off; off += (size_t)NNODES * HEADS;
    float* den  = ws + off; off += (size_t)NNODES * HEADS;
    _Float16* wFrag = (_Float16*)(ws + off); off += 32768; // 128 KB frag buffer

    float* outLink = (float*)d_out;
    float* outExpr = (float*)d_out + NLAB;

    auto gemm = [&](const float* A, const float* Bw, const float* bias, float* C,
                    int rows, int K, int M, int relu) {
        int tilesM  = (M + 15) / 16;
        int mChunks = (tilesM + MT_PER_BLOCK - 1) / MT_PER_BLOCK;
        int rowBlks = (rows + 127) / 128;
        int K32 = K / 32;
        // pre-convert weights into padded fragment-ordered f16 chunks
        int total = mChunks * (K32 << 12);
        wfrag_kernel<<<(total + 255) / 256, 256, 0, stream>>>(Bw, wFrag, K32, M, total);
        dim3 grid((unsigned)rowBlks, (unsigned)mChunks);
        size_t lds = (size_t)K32 * 16384;   // (K32*8*512 + 128*K) f16 * 2B
        switch (K32) {
            case 1:  gemm_wmma_kernel<1><<<grid, 256, lds, stream>>>(wFrag, A, bias, C, rows, M, relu); break;
            case 2:  gemm_wmma_kernel<2><<<grid, 256, lds, stream>>>(wFrag, A, bias, C, rows, M, relu); break;
            default: gemm_wmma_kernel<3><<<grid, 256, lds, stream>>>(wFrag, A, bias, C, rows, M, relu); break;
        }
    };
    auto fill = [&](float* p, long n, float v) {
        fill_kernel<<<(unsigned)((n + 255) / 256), 256, 0, stream>>>(p, n, v);
    };

    const int EH = NEDGES * HEADS, NH = NNODES * HEADS;
    const int ehB = (EH + 255) / 256, nhB = (NH + 255) / 256;

    // -------- GAT layer 0: in 64 -> 4 heads x 64 --------
    gemm(x, W0, nullptr, bufA, NNODES, 64, 256, 0);                       // h0
    alpha_kernel<<<nhB, 256, 0, stream>>>(bufA, as0, ad0, aS, aD, 64);
    fill(mMax, NH, -3.0e38f);
    fill(den,  NH, 0.0f);
    fill(bufB, (long)NNODES * 256, 0.0f);
    edge_max_kernel<<<ehB, 256, 0, stream>>>(src, dst, aS, aD, mMax);
    edge_den_kernel<<<ehB, 256, 0, stream>>>(src, dst, aS, aD, mMax, den);
    edge_agg_kernel<<<ehB, 256, 0, stream>>>(src, dst, aS, aD, mMax, den, bufA, bufB, 64);
    mean_kernel<<<(NNODES * 64 + 255) / 256, 256, 0, stream>>>(bufB, b0, bufC, 64); // hr0

    // -------- GAT layer 1: in 64 -> 4 heads x 32 --------
    gemm(bufC, W1, nullptr, bufA, NNODES, 64, 128, 0);                    // h1
    alpha_kernel<<<nhB, 256, 0, stream>>>(bufA, as1, ad1, aS, aD, 32);
    fill(mMax, NH, -3.0e38f);
    fill(den,  NH, 0.0f);
    fill(bufB, (long)NNODES * 128, 0.0f);
    edge_max_kernel<<<ehB, 256, 0, stream>>>(src, dst, aS, aD, mMax);
    edge_den_kernel<<<ehB, 256, 0, stream>>>(src, dst, aS, aD, mMax, den);
    edge_agg_kernel<<<ehB, 256, 0, stream>>>(src, dst, aS, aD, mMax, den, bufA, bufB, 32);
    mean_kernel<<<(NNODES * 32 + 255) / 256, 256, 0, stream>>>(bufB, b1, bufC, 32); // hr1

    // -------- z = hr1 @ lin_w + lin_b --------
    gemm(bufC, linw, linb, bufA, NNODES, 32, 32, 0);                      // z

    // -------- link decoder --------
    link_kernel<<<(NLAB + 255) / 256, 256, 0, stream>>>(eli, bufA, outLink);

    // -------- expression MLP: 32 -> 32 -> 64 -> 96 -> 500 --------
    gemm(bufA, d1w, d1b, bufC, NNODES, 32, 32, 1);                        // e1
    gemm(bufC, d2w, d2b, bufB, NNODES, 32, 64, 1);                        // e2
    gemm(bufB, d3w, d3b, bufC, NNODES, 64, 96, 1);                        // e3
    gemm(bufC, d4w, d4b, outExpr, NNODES, 96, 500, 0);                    // expr
}
